// LSTM_94489281150
// MI455X (gfx1250) — compile-verified, matrix operand order checked by reading, not run on hardware
//
#include <hip/hip_runtime.h>

// ---------------------------------------------------------------------------
// LSTM LM forward for MI455X (gfx1250), bf16 WMMA + persistent recurrence.
//   T=512, B=32, H=1024, NCLASS=32000, NLAYERS=2 (same weights, same xt).
//   1) one-time converts: w1 -> WxT/WhT bf16 [4H,H] (transposed), fc_w bf16,
//      emb-gather bf16.
//   2) Gx = xe @ Wx for ALL timesteps in one WMMA GEMM (4 N-tiles/wave,
//      software-pipelined B prefetch).
//   3) ONE persistent kernel (16 co-resident WGs) runs all 1023 dependent
//      cell steps with an agent-scope atomic grid barrier. h_prev staged in
//      LDS; c lives in REGISTERS for the whole sequence; Gx prefetched at
//      step head; K-loop software-pipelined.
//   4) final FC y = h @ fc_w^T + fc_b via WMMA.
// ---------------------------------------------------------------------------

typedef __attribute__((ext_vector_type(16))) __bf16 v16bf;
typedef __attribute__((ext_vector_type(8)))  float  v8f;

#define TT     512
#define BB     32
#define HH     1024
#define FOURH  4096
#define NCLS   32000
#define NSTEPS 1023           // 1 + 511*2 dependent cell evaluations
#define SEQ_BLOCKS 16

union Frag { uint4 u[2]; v16bf v; };

__device__ __forceinline__ unsigned short f2bf(float f) {
    unsigned int u = __float_as_uint(f);
    u += 0x7FFFu + ((u >> 16) & 1u);       // round-to-nearest-even
    return (unsigned short)(u >> 16);
}

// Per-lane A/B fragment: two contiguous 8xbf16 (16B) runs.
// ISA 16-bit operand layout: lane<16 -> K {kk..kk+7, kk+16..kk+23},
//                            lane>=16 -> K {kk+8..kk+15, kk+24..kk+31}.
__device__ __forceinline__ v16bf load_frag(const unsigned short* __restrict__ base,
                                           int row, int kk, int half) {
    Frag f;
    const unsigned short* p = base + (size_t)row * HH + kk + half * 8;
    f.u[0] = *reinterpret_cast<const uint4*>(p);
    f.u[1] = *reinterpret_cast<const uint4*>(p + 16);
    return f.v;
}

__device__ __forceinline__ float sigmoidf(float z) {
    return 1.0f / (1.0f + __expf(-z));
}

// ---------------------------------------------------------------------------
// One-time prep kernels
// ---------------------------------------------------------------------------
__global__ void conv_w1_kernel(const float* __restrict__ w1,
                               unsigned short* __restrict__ wxT,
                               unsigned short* __restrict__ whT) {
    int tid = blockIdx.x * blockDim.x + threadIdx.x;   // over 4H*H
    int n = tid >> 10;            // 0..4095 (gate*H + j)
    int k = tid & 1023;
    int g = n >> 10;
    int j = n & 1023;
    size_t src = (size_t)k * 8192 + (size_t)(2 * g) * 1024 + j;
    wxT[(size_t)n * HH + k] = f2bf(w1[src]);
    whT[(size_t)n * HH + k] = f2bf(w1[src + 1024]);
}

__global__ void conv_fcw_kernel(const float* __restrict__ fcw,
                                unsigned short* __restrict__ fcw16) {
    size_t tid = (size_t)blockIdx.x * blockDim.x + threadIdx.x; // NCLS*H
    fcw16[tid] = f2bf(fcw[tid]);
}

__global__ void gather_kernel(const int* __restrict__ x,
                              const float* __restrict__ emb,
                              unsigned short* __restrict__ xe16) {
    size_t tid = (size_t)blockIdx.x * blockDim.x + threadIdx.x; // T*B*H
    size_t row = tid >> 10;       // t*B + b
    int    col = (int)(tid & 1023);
    int idx = x[row];
    xe16[tid] = f2bf(emb[(size_t)idx * HH + col]);
}

__global__ void init_state_kernel(unsigned short* __restrict__ h16,
                                  unsigned int* __restrict__ bar) {
    int tid = blockIdx.x * blockDim.x + threadIdx.x;   // B*H
    h16[tid] = 0;
    if (tid < 2) bar[tid] = 0u;   // {count, generation}
}

// ---------------------------------------------------------------------------
// Gx = xe @ Wx : [16384,1024] x [1024,4096] -> f32. Each wave: 4 N-tiles
// sharing one A fragment; B fragments double-buffered so WMMAs overlap the
// next chunk's loads (avoids s_wait_loadcnt 0x0 full drains).
// ---------------------------------------------------------------------------
__global__ void gemm_gx_kernel(const unsigned short* __restrict__ xe16,
                               const unsigned short* __restrict__ wxT,
                               float* __restrict__ gx) {
    int wave = threadIdx.x >> 5, lane = threadIdx.x & 31;
    int gw = blockIdx.x * 8 + wave;       // 0 .. 65535
    int nt4 = gw & 63;                    // 64 groups of 4 N-tiles
    int mt  = gw >> 6;                    // 1024 M tiles
    int m0 = mt << 4, n0 = nt4 << 6;      // 64-wide N slab
    int half = lane >> 4, r = lane & 15;

    v8f acc0 = {}, acc1 = {}, acc2 = {}, acc3 = {};
    v16bf a  = load_frag(xe16, m0 + r, 0, half);
    v16bf b0 = load_frag(wxT, n0 +  0 + r, 0, half);
    v16bf b1 = load_frag(wxT, n0 + 16 + r, 0, half);
    v16bf b2 = load_frag(wxT, n0 + 32 + r, 0, half);
    v16bf b3 = load_frag(wxT, n0 + 48 + r, 0, half);
#pragma unroll
    for (int kk = 0; kk < HH; kk += 32) {
        v16bf na, nb0, nb1, nb2, nb3;
        if (kk + 32 < HH) {
            na  = load_frag(xe16, m0 + r, kk + 32, half);
            nb0 = load_frag(wxT, n0 +  0 + r, kk + 32, half);
            nb1 = load_frag(wxT, n0 + 16 + r, kk + 32, half);
            nb2 = load_frag(wxT, n0 + 32 + r, kk + 32, half);
            nb3 = load_frag(wxT, n0 + 48 + r, kk + 32, half);
        }
        acc0 = __builtin_amdgcn_wmma_f32_16x16x32_bf16(false, a, false, b0, (short)0, acc0, false, false);
        acc1 = __builtin_amdgcn_wmma_f32_16x16x32_bf16(false, a, false, b1, (short)0, acc1, false, false);
        acc2 = __builtin_amdgcn_wmma_f32_16x16x32_bf16(false, a, false, b2, (short)0, acc2, false, false);
        acc3 = __builtin_amdgcn_wmma_f32_16x16x32_bf16(false, a, false, b3, (short)0, acc3, false, false);
        a = na; b0 = nb0; b1 = nb1; b2 = nb2; b3 = nb3;
    }
#pragma unroll
    for (int v = 0; v < 8; ++v) {
        int m = m0 + half * 8 + v;
        float* row = gx + (size_t)m * FOURH + n0 + r;
        row[ 0] = acc0[v];
        row[16] = acc1[v];
        row[32] = acc2[v];
        row[48] = acc3[v];
    }
}

// ---------------------------------------------------------------------------
// Agent-scope sense-reversing grid barrier (16 co-resident WGs).
// ---------------------------------------------------------------------------
__device__ __forceinline__ void grid_barrier(unsigned int* __restrict__ bar,
                                             unsigned int my_gen) {
    __threadfence();                        // release this thread's h stores
    __syncthreads();
    if (threadIdx.x == 0) {
        unsigned int arrived =
            __hip_atomic_fetch_add(&bar[0], 1u, __ATOMIC_ACQ_REL, __HIP_MEMORY_SCOPE_AGENT);
        if (arrived == SEQ_BLOCKS - 1) {
            __hip_atomic_store(&bar[0], 0u, __ATOMIC_RELAXED, __HIP_MEMORY_SCOPE_AGENT);
            __hip_atomic_fetch_add(&bar[1], 1u, __ATOMIC_RELEASE, __HIP_MEMORY_SCOPE_AGENT);
        } else {
            while (__hip_atomic_load(&bar[1], __ATOMIC_ACQUIRE, __HIP_MEMORY_SCOPE_AGENT)
                   == my_gen) {
                __builtin_amdgcn_s_sleep(1);
            }
        }
    }
    __syncthreads();
    __threadfence();                        // acquire-side for vector loads
}

// ---------------------------------------------------------------------------
// Persistent recurrence: all 1023 cell steps in one launch.
// 16 WGs x 8 waves = 128 waves; wave owns (m-tile, n-tile) and all 4 gates;
// its 8 c values stay in registers for the whole sequence.
// Step s: s==0 -> (t=0, write); s odd -> (t=(s+1)/2, no write);
//         s even -> (t=s/2, write).
// ---------------------------------------------------------------------------
__global__ void lstm_seq_kernel(unsigned short* __restrict__ hA,   // zero-inited
                                unsigned short* __restrict__ hB,
                                const unsigned short* __restrict__ whT,
                                const float* __restrict__ gx,
                                const float* __restrict__ bias,    // [4H]
                                float* __restrict__ out_h,
                                unsigned int* __restrict__ bar) {
    __shared__ unsigned short hs[BB * HH];        // 64 KB of the 320 KB/WGP LDS

    int wave = threadIdx.x >> 5, lane = threadIdx.x & 31;
    int gw = blockIdx.x * 8 + wave;       // 0..127 (fixed across steps)
    int mt = gw & 1;
    int nt = gw >> 1;                     // 0..63
    int m0 = mt << 4, n0 = nt << 4;
    int half = lane >> 4, r = lane & 15;
    int j = n0 + r;

    // bias is constant per lane position across all steps
    float bias4[4];
#pragma unroll
    for (int g = 0; g < 4; ++g) bias4[g] = bias[g * HH + j];

    // cell state lives in registers: (b=m0+8*half+v, j) owned by this lane
    float creg[8];
#pragma unroll
    for (int v = 0; v < 8; ++v) creg[v] = 0.0f;

    const unsigned short* hprev = hA;
    unsigned short*       hnext = hB;

#pragma unroll 1
    for (int s = 0; s < NSTEPS; ++s) {
        int t, wt;
        if (s == 0)      { t = 0;            wt = 0;  }
        else if (s & 1)  { t = (s + 1) >> 1; wt = -1; }
        else             { t = s >> 1;       wt = t;  }

        // Prefetch this step's Gx values (independent of the GEMM below).
        const float* gxt = gx + (size_t)t * BB * FOURH;
        float gxv[4][8];
#pragma unroll
        for (int v = 0; v < 8; ++v) {
            const float* grow = gxt + (size_t)(m0 + half * 8 + v) * FOURH + j;
#pragma unroll
            for (int g = 0; g < 4; ++g) gxv[g][v] = grow[g * HH];
        }

        // Stage h_prev (64 KB bf16) into LDS: 4 waves share each A fragment.
        for (int i = threadIdx.x; i < (BB * HH) / 8; i += blockDim.x)
            reinterpret_cast<uint4*>(hs)[i] =
                reinterpret_cast<const uint4*>(hprev)[i];
        __syncthreads();

        v8f acc0 = {}, acc1 = {}, acc2 = {}, acc3 = {};
        v16bf b0 = load_frag(whT, 0 * HH + n0 + r, 0, half);
        v16bf b1 = load_frag(whT, 1 * HH + n0 + r, 0, half);
        v16bf b2 = load_frag(whT, 2 * HH + n0 + r, 0, half);
        v16bf b3 = load_frag(whT, 3 * HH + n0 + r, 0, half);
#pragma unroll
        for (int kk = 0; kk < HH; kk += 32) {
            v16bf a = load_frag(hs, m0 + r, kk, half);   // LDS, low latency
            v16bf nb0, nb1, nb2, nb3;
            if (kk + 32 < HH) {
                nb0 = load_frag(whT, 0 * HH + n0 + r, kk + 32, half);
                nb1 = load_frag(whT, 1 * HH + n0 + r, kk + 32, half);
                nb2 = load_frag(whT, 2 * HH + n0 + r, kk + 32, half);
                nb3 = load_frag(whT, 3 * HH + n0 + r, kk + 32, half);
            }
            acc0 = __builtin_amdgcn_wmma_f32_16x16x32_bf16(false, a, false, b0, (short)0, acc0, false, false);
            acc1 = __builtin_amdgcn_wmma_f32_16x16x32_bf16(false, a, false, b1, (short)0, acc1, false, false);
            acc2 = __builtin_amdgcn_wmma_f32_16x16x32_bf16(false, a, false, b2, (short)0, acc2, false, false);
            acc3 = __builtin_amdgcn_wmma_f32_16x16x32_bf16(false, a, false, b3, (short)0, acc3, false, false);
            b0 = nb0; b1 = nb1; b2 = nb2; b3 = nb3;
        }

#pragma unroll
        for (int v = 0; v < 8; ++v) {
            int b = m0 + half * 8 + v;          // D layout: row = 8*(lane/16)+v
            float zct = acc0[v] + gxv[0][v] + bias4[0];
            float zi  = acc1[v] + gxv[1][v] + bias4[1];
            float zf  = acc2[v] + gxv[2][v] + bias4[2];
            float zo  = acc3[v] + gxv[3][v] + bias4[3];
            float ctilde = tanhf(zct);
            float it = sigmoidf(zi);
            float ft = sigmoidf(zf);
            float cn = it * ctilde + ft * creg[v];
            creg[v] = cn;
            float hn = zo * tanhf(cn);          // faithful: no sigmoid on o-gate
            hnext[(size_t)b * HH + j] = f2bf(hn);
            if (wt >= 0)
                out_h[(size_t)b * (TT * HH) + (size_t)wt * HH + j] = hn;
        }

        grid_barrier(bar, (unsigned int)s);     // also guards LDS restage

        const unsigned short* tmp = hprev;
        hprev = hnext;
        hnext = const_cast<unsigned short*>(tmp);
    }
}

// ---------------------------------------------------------------------------
// y = h_final @ fc_w^T + fc_b : [32,1024] x [1024,32000]
// ---------------------------------------------------------------------------
__global__ void gemm_fc_kernel(const unsigned short* __restrict__ h16,
                               const unsigned short* __restrict__ fcw16,
                               const float* __restrict__ fcb,
                               float* __restrict__ y) {
    int wave = threadIdx.x >> 5, lane = threadIdx.x & 31;
    int gw = blockIdx.x * 8 + wave;       // 0..3999
    int mt = gw & 1;
    int nt = gw >> 1;                     // 0..1999
    int m0 = mt << 4, n0 = nt << 4;
    int half = lane >> 4, r = lane & 15;

    v8f acc = {};
    v16bf a = load_frag(h16,   m0 + r, 0, half);
    v16bf b = load_frag(fcw16, n0 + r, 0, half);
#pragma unroll
    for (int kk = 0; kk < HH; kk += 32) {
        v16bf na, nb;
        if (kk + 32 < HH) {
            na = load_frag(h16,   m0 + r, kk + 32, half);
            nb = load_frag(fcw16, n0 + r, kk + 32, half);
        }
        acc = __builtin_amdgcn_wmma_f32_16x16x32_bf16(false, a, false, b,
                                                      (short)0, acc, false, false);
        a = na; b = nb;
    }
    int n = n0 + r;
    float bn = fcb[n];
#pragma unroll
    for (int v = 0; v < 8; ++v) {
        int m = m0 + half * 8 + v;
        y[(size_t)m * NCLS + n] = acc[v] + bn;
    }
}

// ---------------------------------------------------------------------------
extern "C" void kernel_launch(void* const* d_in, const int* in_sizes, int n_in,
                              void* d_out, int out_size, void* d_ws, size_t ws_size,
                              hipStream_t stream) {
    const int*   x    = (const int*)  d_in[0];
    const float* emb  = (const float*)d_in[1];
    const float* w1   = (const float*)d_in[2];
    const float* b1   = (const float*)d_in[3];
    const float* fc_w = (const float*)d_in[4];
    const float* fc_b = (const float*)d_in[5];

    float* y     = (float*)d_out;                       // [B, NCLS]
    float* out_h = y + (size_t)BB * NCLS;               // [B, T*H]

    char* ws = (char*)d_ws;
    auto carve = [&](size_t bytes) -> char* {
        char* p = ws;
        ws += (bytes + 255) & ~(size_t)255;
        return p;
    };
    unsigned short* wxT   = (unsigned short*)carve((size_t)FOURH * HH * 2);
    unsigned short* whT   = (unsigned short*)carve((size_t)FOURH * HH * 2);
    unsigned short* fcw16 = (unsigned short*)carve((size_t)NCLS * HH * 2);
    unsigned short* xe16  = (unsigned short*)carve((size_t)TT * BB * HH * 2);
    float*          gx    = (float*)         carve((size_t)TT * BB * FOURH * 4);
    unsigned short* hbuf0 = (unsigned short*)carve((size_t)BB * HH * 2);
    unsigned short* hbuf1 = (unsigned short*)carve((size_t)BB * HH * 2);
    unsigned int*   bar   = (unsigned int*)  carve(256);

    // 1) one-time converts + gather + state/barrier init
    conv_w1_kernel  <<<(FOURH * HH) / 256, 256, 0, stream>>>(w1, wxT, whT);
    conv_fcw_kernel <<<((size_t)NCLS * HH) / 256, 256, 0, stream>>>(fc_w, fcw16);
    gather_kernel   <<<((size_t)TT * BB * HH) / 256, 256, 0, stream>>>(x, emb, xe16);
    init_state_kernel<<<(BB * HH) / 256, 256, 0, stream>>>(hbuf0, bar);

    // 2) all-timestep input GEMM Gx = xe @ Wx
    gemm_gx_kernel<<<8192, 256, 0, stream>>>(xe16, wxT, gx);

    // 3) entire recurrence in ONE persistent launch (16 co-resident WGs)
    lstm_seq_kernel<<<SEQ_BLOCKS, 256, 0, stream>>>(hbuf0, hbuf1, whT, gx, b1,
                                                    out_h, bar);

    // 4) classifier; after 1023 steps (odd #swaps) the final h is in hbuf1
    gemm_fc_kernel<<<500, 256, 0, stream>>>(hbuf1, fcw16, fc_b, y);
}
